// Net_51977694216541
// MI455X (gfx1250) — matrix operand order repaired and checked by reading, not compiled, not run on hardware
//
#include <hip/hip_runtime.h>
#include <stdint.h>

typedef __attribute__((ext_vector_type(16))) _Float16 v16h;
typedef __attribute__((ext_vector_type(8)))  float    v8f;
typedef __attribute__((ext_vector_type(4)))  _Float16 h4;

#define N_NODES 100000
#define N_EDGES 300000
#define IN_C 128
#define HID 256
#define OUT_C 64
#define LDC 512   // combined [mean | x] row length (halves)

// ---------------- utility kernels ----------------

__global__ void k_f32_to_f16(const float* __restrict__ src, _Float16* __restrict__ dst, int n) {
  int i = blockIdx.x * blockDim.x + threadIdx.x;
  if (i < n) dst[i] = (_Float16)src[i];
}

// Pack f32 row-major weights [K x Nout] into WMMA-B fragment-ready layout.
// Per-fragment layout is [lane][v] (8 contiguous u32 per lane -> 2x b128 loads):
//   dst[(ktGlobal*NT + nt)*256 + lane*8 + v] = half2{ W[k][n], W[k+1][n] }
// B k order used here (contiguous, same as A): lanes 0-15 hold K=0..15 (VGPRv = K=2v,2v+1), lanes 16-31 K=16..31.
__global__ void k_pack_weights(const float* __restrict__ W, uint32_t* __restrict__ dst,
                               int Nout, int ktOff, int total) {
  int tid = blockIdx.x * blockDim.x + threadIdx.x;
  if (tid >= total) return;
  int v    = tid & 7;
  int lane = (tid >> 3) & 31;
  int rest = tid >> 8;
  int NT   = Nout >> 4;
  int nt   = rest % NT;
  int ktL  = rest / NT;
  int n = (nt << 4) + (lane & 15);
  int k = (ktL << 5) + (v << 1) + ((lane >> 4) << 4);
  _Float16 h0 = (_Float16)W[(size_t)k * Nout + n];
  _Float16 h1 = (_Float16)W[(size_t)(k + 1) * Nout + n];
  uint32_t u = (uint32_t)__builtin_bit_cast(unsigned short, h0)
             | ((uint32_t)__builtin_bit_cast(unsigned short, h1) << 16);
  dst[(size_t)((ktOff + ktL) * NT + nt) * 256 + lane * 8 + v] = u;
}

__global__ void k_count(const int* __restrict__ dstIdx, float* __restrict__ cnt, int nE) {
  int e = blockIdx.x * blockDim.x + threadIdx.x;
  if (e < nE) atomicAdd(&cnt[dstIdx[e]], 1.0f);
}

// 64 threads per edge, 4 channels each: gather f16 x-row (x half of combined
// buffer, row stride LDC), f32 atomic-add into the L2-resident accumulator.
__global__ void k_scatter(const int* __restrict__ srcIdx, const int* __restrict__ dstIdx,
                          const _Float16* __restrict__ xH, float* __restrict__ agg, int nE) {
  int tid = blockIdx.x * blockDim.x + threadIdx.x;
  int e = tid >> 6;
  if (e >= nE) return;
  int c = (tid & 63) << 2;
  int s = srcIdx[e];
  int d = dstIdx[e];
  h4 hv = *reinterpret_cast<const h4*>(xH + (size_t)s * LDC + c);
  float* q = agg + (size_t)d * HID + c;
  atomicAdd(q + 0, (float)hv.x);
  atomicAdd(q + 1, (float)hv.y);
  atomicAdd(q + 2, (float)hv.z);
  atomicAdd(q + 3, (float)hv.w);
}

// writes the mean half of the combined buffer (row stride LDC)
__global__ void k_finalize_mean(const float* __restrict__ agg, const float* __restrict__ cnt,
                                _Float16* __restrict__ meanH, int n) {
  int i = blockIdx.x * blockDim.x + threadIdx.x;
  if (i >= n) return;
  float c = cnt[i >> 8];               // HID == 256
  meanH[(size_t)(i >> 8) * LDC + (i & 255)] = (_Float16)(agg[i] / fmaxf(c, 1.0f));
}

// ---------------- WMMA GEMM ----------------
// C[rows x Nout] = A[rows x K] (f16, row-major, ld=lda) @ Wp (f16 packed) + bias.
// One wave computes a 16x64 tile (4 n-tiles). Double-buffered K loop (KT even),
// branch-free prefetch one step ahead so the waitcnt pass can emit partial waits.
// mode: 0 = stage0 (inp f32 + relu->f16), 1 = layer (relu + 0.2*resid -> f16),
//       2 = raw f32 out.

union FragH { v16h h; uint32_t u[8]; };

__device__ __forceinline__ void load_afrag(FragH& f, const _Float16* __restrict__ rowp) {
  // A layout (matches packed B): lanes0-15: v0..7 -> K0..15; lanes16-31 +16.
#pragma unroll
  for (int v = 0; v < 8; ++v) {
    int k = v << 1;
    f.u[v] = *reinterpret_cast<const uint32_t*>(rowp + k);
  }
}

__device__ __forceinline__ void load_bfrags(FragH* __restrict__ f,
                                            const uint32_t* __restrict__ wbase) {
#pragma unroll
  for (int t = 0; t < 4; ++t)
#pragma unroll
    for (int v = 0; v < 8; ++v) f[t].u[v] = wbase[t * 256 + v];
}

__global__ void k_gemm_wmma(const _Float16* __restrict__ A, int lda, int KT,
                            const uint32_t* __restrict__ Wp, const float* __restrict__ bias,
                            int Nout, int mode,
                            const float* __restrict__ resid,
                            float* __restrict__ outF, int ldF,
                            _Float16* __restrict__ outH, int ldH,
                            int totalTiles) {
  int wave = (blockIdx.x * blockDim.x + threadIdx.x) >> 5;
  int lane = threadIdx.x & 31;
  if (wave >= totalTiles) return;          // wave-uniform: EXEC all-ones for WMMA

  int colGroups = Nout >> 6;
  int tileM   = wave / colGroups;
  int colBase = (wave % colGroups) << 6;
  int rowBase = tileM << 4;
  int NT  = Nout >> 4;
  int nt0 = colBase >> 4;
  int lhalf = lane >> 4;
  int m = lane & 15;

  const _Float16* ap = A + (size_t)(rowBase + m) * lda + (lhalf << 4);
  const uint32_t* wp = Wp + (size_t)nt0 * 256 + (size_t)lane * 8;
  const size_t wstep = (size_t)NT * 256;

  v8f acc[4] = {};
  FragH fa0, fb0[4], fa1, fb1[4];

  // prologue: fragments for kt = 0
  load_afrag(fa0, ap);
  load_bfrags(fb0, wp);

  for (int kt = 0; kt < KT; kt += 2) {
    // prefetch kt+1 (always valid: KT even)
    load_afrag(fa1, ap + (size_t)(kt + 1) * 32);
    load_bfrags(fb1, wp + (size_t)(kt + 1) * wstep);
#pragma unroll
    for (int t = 0; t < 4; ++t)
      acc[t] = __builtin_amdgcn_wmma_f32_16x16x32_f16(
          false, fa0.h, false, fb0[t].h, (short)0, acc[t], false, false);
    // prefetch kt+2 (over-reads one tile past the end on the last step; the
    // workspace is padded so this is harmless and the data is never used)
    load_afrag(fa0, ap + (size_t)(kt + 2) * 32);
    load_bfrags(fb0, wp + (size_t)(kt + 2) * wstep);
#pragma unroll
    for (int t = 0; t < 4; ++t)
      acc[t] = __builtin_amdgcn_wmma_f32_16x16x32_f16(
          false, fa1.h, false, fb1[t].h, (short)0, acc[t], false, false);
  }

#pragma unroll
  for (int t = 0; t < 4; ++t) {
    int col = colBase + (t << 4) + m;
    float b = bias[col];
#pragma unroll
    for (int v = 0; v < 8; ++v) {
      int row = rowBase + v + (lhalf << 3);
      float val = acc[t][v] + b;
      if (mode == 0) {
        outF[(size_t)row * ldF + col] = val;                       // inp (pre-relu)
        outH[(size_t)row * ldH + col] = (_Float16)fmaxf(val, 0.0f);
      } else if (mode == 1) {
        float r = fmaxf(val, 0.0f) + 0.2f * resid[(size_t)row * HID + col];
        outH[(size_t)row * ldH + col] = (_Float16)r;
      } else {
        outF[(size_t)row * ldF + col] = val;
      }
    }
  }
}

// one wave per node, 64 logits (2 per lane)
__global__ void k_log_softmax64(float* __restrict__ io, int nNodes) {
  int w = (blockIdx.x * blockDim.x + threadIdx.x) >> 5;
  int lane = threadIdx.x & 31;
  if (w >= nNodes) return;
  float* p = io + (size_t)w * OUT_C;
  float a = p[lane];
  float b = p[lane + 32];
  float mx = fmaxf(a, b);
#pragma unroll
  for (int off = 16; off > 0; off >>= 1) mx = fmaxf(mx, __shfl_xor(mx, off, 32));
  float s = expf(a - mx) + expf(b - mx);
#pragma unroll
  for (int off = 16; off > 0; off >>= 1) s += __shfl_xor(s, off, 32);
  float ls = logf(s);
  p[lane]      = a - mx - ls;
  p[lane + 32] = b - mx - ls;
}

// ---------------- host launcher ----------------

extern "C" void kernel_launch(void* const* d_in, const int* in_sizes, int n_in,
                              void* d_out, int out_size, void* d_ws, size_t ws_size,
                              hipStream_t stream) {
  (void)in_sizes; (void)n_in; (void)out_size; (void)ws_size;
  const float* x    = (const float*)d_in[0];
  const int*   ei   = (const int*)d_in[1];
  const float* W_in = (const float*)d_in[2];
  const float* b_in = (const float*)d_in[3];
  const float* Wl1  = (const float*)d_in[4];
  const float* bl1  = (const float*)d_in[5];
  const float* Wr1  = (const float*)d_in[6];
  const float* Wl2  = (const float*)d_in[7];
  const float* bl2  = (const float*)d_in[8];
  const float* Wr2  = (const float*)d_in[9];
  const float* Wl3  = (const float*)d_in[10];
  const float* bl3  = (const float*)d_in[11];
  const float* Wr3  = (const float*)d_in[12];
  const int* srcIdx = ei;
  const int* dstIdx = ei + N_EDGES;

  char* ws = (char*)d_ws;
  size_t off = 0;
  auto take = [&](size_t bytes) -> char* {
    char* p = ws + off;
    off += (bytes + 255) & ~(size_t)255;
    return p;
  };

  _Float16* xinH  = (_Float16*)take((size_t)N_NODES * IN_C * 2);
  float*    inp   = (float*)   take((size_t)N_NODES * HID * 4);
  _Float16* combA = (_Float16*)take((size_t)N_NODES * LDC * 2);  // [mean | x]
  _Float16* combB = (_Float16*)take((size_t)N_NODES * LDC * 2);  // [mean | x]
  float*    agg   = (float*)   take((size_t)N_NODES * HID * 4);
  float*    cnt   = (float*)   take((size_t)N_NODES * 4);
  uint32_t* WpIn  = (uint32_t*)take((size_t)IN_C * HID / 2 * 4);
  uint32_t* Wp1   = (uint32_t*)take((size_t)2 * HID * HID / 2 * 4);
  uint32_t* Wp2   = (uint32_t*)take((size_t)2 * HID * HID / 2 * 4);
  uint32_t* Wp3   = (uint32_t*)take((size_t)2 * HID * OUT_C / 2 * 4);
  (void)take(64 * 1024);                 // guard for end-of-loop prefetch over-read
  float* outPre = (float*)d_out;

  // ---- prep: convert input + pack all weights into WMMA-B layout ----
  {
    int n = N_NODES * IN_C;
    k_f32_to_f16<<<(n + 255) / 256, 256, 0, stream>>>(x, xinH, n);
  }
  {
    int tot = IN_C * HID / 2;
    k_pack_weights<<<(tot + 255) / 256, 256, 0, stream>>>(W_in, WpIn, HID, 0, tot);
  }
  {
    int tot = HID * HID / 2;
    k_pack_weights<<<(tot + 255) / 256, 256, 0, stream>>>(Wl1, Wp1, HID, 0,        tot);
    k_pack_weights<<<(tot + 255) / 256, 256, 0, stream>>>(Wr1, Wp1, HID, HID / 32, tot);
    k_pack_weights<<<(tot + 255) / 256, 256, 0, stream>>>(Wl2, Wp2, HID, 0,        tot);
    k_pack_weights<<<(tot + 255) / 256, 256, 0, stream>>>(Wr2, Wp2, HID, HID / 32, tot);
  }
  {
    int tot = HID * OUT_C / 2;
    k_pack_weights<<<(tot + 255) / 256, 256, 0, stream>>>(Wl3, Wp3, OUT_C, 0,        tot);
    k_pack_weights<<<(tot + 255) / 256, 256, 0, stream>>>(Wr3, Wp3, OUT_C, HID / 32, tot);
  }
  hipMemsetAsync(cnt, 0, (size_t)N_NODES * 4, stream);
  k_count<<<(N_EDGES + 255) / 256, 256, 0, stream>>>(dstIdx, cnt, N_EDGES);

  // ---- stage 0: inp = x @ W_in + b_in ; combA.x = relu(inp) ----
  int tiles256 = (N_NODES / 16) * (HID / 64);   // 25000 waves
  k_gemm_wmma<<<(tiles256 + 7) / 8, 256, 0, stream>>>(
      xinH, IN_C, IN_C / 32, WpIn, b_in, HID, 0, (const float*)nullptr,
      inp, HID, combA + HID, LDC, tiles256);

  // ---- layers 1 & 2 (ping-pong combined buffers) ----
  const float* biases[2] = { bl1, bl2 };
  const uint32_t* Wps[2] = { Wp1, Wp2 };
  _Float16* combs[3] = { combA, combB, combA };
  for (int L = 0; L < 2; ++L) {
    hipMemsetAsync(agg, 0, (size_t)N_NODES * HID * 4, stream);
    int st = N_EDGES * 64;
    k_scatter<<<(st + 255) / 256, 256, 0, stream>>>(srcIdx, dstIdx, combs[L] + HID, agg, N_EDGES);
    int n = N_NODES * HID;
    k_finalize_mean<<<(n + 255) / 256, 256, 0, stream>>>(agg, cnt, combs[L], n);
    k_gemm_wmma<<<(tiles256 + 7) / 8, 256, 0, stream>>>(
        combs[L], LDC, LDC / 32, Wps[L], biases[L], HID, 1, inp,
        (float*)nullptr, 0, combs[L + 1] + HID, LDC, tiles256);
  }

  // ---- layer 3: K=512 -> 64, raw f32 into d_out ----
  hipMemsetAsync(agg, 0, (size_t)N_NODES * HID * 4, stream);
  {
    int st = N_EDGES * 64;
    k_scatter<<<(st + 255) / 256, 256, 0, stream>>>(srcIdx, dstIdx, combA + HID, agg, N_EDGES);
    int n = N_NODES * HID;
    k_finalize_mean<<<(n + 255) / 256, 256, 0, stream>>>(agg, cnt, combA, n);
  }
  int tiles64 = (N_NODES / 16) * (OUT_C / 64);  // 6250 waves
  k_gemm_wmma<<<(tiles64 + 7) / 8, 256, 0, stream>>>(
      combA, LDC, LDC / 32, Wp3, bl3, OUT_C, 2, (const float*)nullptr,
      outPre, OUT_C, (_Float16*)nullptr, 0, tiles64);

  // ---- log_softmax over 64 logits, in place on d_out ----
  k_log_softmax64<<<(N_NODES + 7) / 8, 256, 0, stream>>>(outPre, N_NODES);
}
